// EquivariantTransformerBlock_12197707121075
// MI455X (gfx1250) — matrix operand order repaired and checked by reading, not compiled, not run on hardware
//
#include <hip/hip_runtime.h>
#include <hip/hip_bf16.h>
#include <math.h>

// ---------------------------------------------------------------------------
// Problem constants (from reference)
// ---------------------------------------------------------------------------
#define NN     10000
#define EE     320000
#define F0C    32
#define F1C    16
#define KC     48        // F0+F1
#define HC     4
#define HIDC   64
#define INV_SQRT3  0.57735026918962576f
#define INV48      (1.0f/48.0f)          // 1/sqrt(FAN_LOGIT)
#define INV48SQ3   (INV48*INV_SQRT3)
#define INVSQ48    0.14433756729740643f  // 1/sqrt(48)

typedef __attribute__((ext_vector_type(16))) _Float16 v16h;
typedef __attribute__((ext_vector_type(8)))  float    v8f;

#define WMMA_F16(a,b,c) __builtin_amdgcn_wmma_f32_16x16x32_f16( \
    false,(a),false,(b),(short)0,(c),false,false)

// f16 16x16x32 A-matrix lane layout (ISA 7.12.2): lane<16 holds M=lane,
// K={0..7,16..23}; lane>=16 holds M=lane-16, K={8..15,24..31}.
__device__ __forceinline__ int a_kidx(int p, int hs) {
  return (p < 4) ? ((hs ? 8 : 0) + 2 * p) : ((hs ? 24 : 16) + 2 * (p - 4));
}
// B-matrix (32x16): same K order as A (lane<16 K={0..7,16..23}, lane>=16 K={8..15,24..31}), N=lane%16.
__device__ __forceinline__ int b_kidx(int p, int hs) {
  return (p < 4) ? ((hs ? 8 : 0) + 2 * p) : ((hs ? 24 : 16) + 2 * (p - 4));
}

__device__ __forceinline__ float gelu_tanh(float x) {
  float x3 = x * x * x;
  return 0.5f * x * (1.0f + tanhf(0.7978845608028654f * (x + 0.044715f * x3)));
}

// ---------------------------------------------------------------------------
// K0: per-edge MLP collapses to a constant (y0 == 1). Compute the 96 key
// gates and 96 value gates once. block 0 -> key, block 1 -> value.
// ---------------------------------------------------------------------------
__global__ __launch_bounds__(128) void k_gates(
    const float* __restrict__ Wk1, const float* __restrict__ Wk2,
    const float* __restrict__ Wk3, const float* __restrict__ Wv1,
    const float* __restrict__ Wv2, const float* __restrict__ Wv3,
    float* __restrict__ gates) {
  const float* W1 = blockIdx.x ? Wv1 : Wk1;
  const float* W2 = blockIdx.x ? Wv2 : Wk2;
  const float* W3 = blockIdx.x ? Wv3 : Wk3;
  float* g = gates + blockIdx.x * 96;
  __shared__ float h1[HIDC];
  __shared__ float h2[HIDC];
  int t = threadIdx.x;
  if (t < HIDC) h1[t] = gelu_tanh(W1[t]);           // fan_in = 1
  __syncthreads();
  if (t < HIDC) {
    float acc = 0.f;
    for (int k = 0; k < HIDC; ++k) acc += h1[k] * W2[k * HIDC + t];
    h2[t] = gelu_tanh(acc * 0.125f);                // 1/sqrt(64)
  }
  __syncthreads();
  if (t < 2 * KC) {
    float acc = 0.f;
    for (int k = 0; k < HIDC; ++k) acc += h2[k] * W3[k * (2 * KC) + t];
    g[t] = acc * 0.125f;
  }
}

// ---------------------------------------------------------------------------
// K1: zero accumulators (m, z, ns, nv are contiguous in workspace)
// ---------------------------------------------------------------------------
__global__ __launch_bounds__(256) void k_zero(float* __restrict__ p, int n) {
  int i = blockIdx.x * blockDim.x + threadIdx.x;
  if (i < n) p[i] = 0.f;
}

// ---------------------------------------------------------------------------
// K2: U[n, j*4+h] = sum_i node_s[n,i] * Wlog0[i,j,h] * gk[j] / 48
// GEMM [10000,32] x [32,192]; one wave per 16x16 tile, single WMMA (K=32).
// ---------------------------------------------------------------------------
__global__ __launch_bounds__(256) void k_precomp_u(
    const float* __restrict__ node_s, const float* __restrict__ Wlog0,
    const float* __restrict__ gates, float* __restrict__ U) {
  int wid = (blockIdx.x * blockDim.x + threadIdx.x) >> 5;
  if (wid >= (NN / 16) * 12) return;
  int lane = threadIdx.x & 31;
  int hs = lane >> 4, lm = lane & 15;
  int rt = wid / 12, ct = wid % 12;
  int m0 = rt * 16, col = ct * 16 + lm;

  v16h a, b;
  const float* arow = node_s + (size_t)(m0 + lm) * F0C;
#pragma unroll
  for (int p = 0; p < 8; ++p) {
    int k = a_kidx(p, hs);
    a[2 * p]     = (_Float16)arow[k];
    a[2 * p + 1] = (_Float16)arow[k + 1];
  }
  float gs = gates[col >> 2] * INV48;   // fold key gate + logit fan norm
#pragma unroll
  for (int p = 0; p < 8; ++p) {
    int k = b_kidx(p, hs);
    b[2 * p]     = (_Float16)(Wlog0[(size_t)k * 192 + col] * gs);
    b[2 * p + 1] = (_Float16)(Wlog0[(size_t)(k + 1) * 192 + col] * gs);
  }
  v8f c = {};
  c = WMMA_F16(a, b, c);
#pragma unroll
  for (int p = 0; p < 8; ++p) {
    int row = p + (hs ? 8 : 0);
    U[(size_t)(m0 + row) * 192 + col] = c[p];
  }
}

// ---------------------------------------------------------------------------
// K3: WV[n, j,c,h] = sum_i node_v[n,i,c] * Wlog1[i,j,h] * gk[48+j]/(48*sqrt3)
// 3 GEMMs (c=0..2) of [10000,16] x [16,192] (K padded to 32 with zeros).
// ---------------------------------------------------------------------------
__global__ __launch_bounds__(256) void k_precomp_wv(
    const float* __restrict__ node_v, const float* __restrict__ Wlog1,
    const float* __restrict__ gates, float* __restrict__ WV) {
  int wid = (blockIdx.x * blockDim.x + threadIdx.x) >> 5;
  if (wid >= (NN / 16) * 12 * 3) return;
  int lane = threadIdx.x & 31;
  int hs = lane >> 4, lm = lane & 15;
  int cc = wid / ((NN / 16) * 12);
  int rem = wid % ((NN / 16) * 12);
  int rt = rem / 12, ct = rem % 12;
  int m0 = rt * 16, col = ct * 16 + lm;

  v16h a, b;
#pragma unroll
  for (int p = 0; p < 8; ++p) {
    int k = a_kidx(p, hs);
    float a0 = (k < F1C)     ? node_v[((size_t)(m0 + lm) * F1C + k) * 3 + cc] : 0.f;
    float a1 = (k + 1 < F1C) ? node_v[((size_t)(m0 + lm) * F1C + k + 1) * 3 + cc] : 0.f;
    a[2 * p]     = (_Float16)a0;
    a[2 * p + 1] = (_Float16)a1;
  }
  float gs = gates[48 + (col >> 2)] * INV48SQ3;
#pragma unroll
  for (int p = 0; p < 8; ++p) {
    int k = b_kidx(p, hs);
    float b0 = (k < F1C)     ? Wlog1[(size_t)k * 192 + col] * gs : 0.f;
    float b1 = (k + 1 < F1C) ? Wlog1[(size_t)(k + 1) * 192 + col] * gs : 0.f;
    b[2 * p]     = (_Float16)b0;
    b[2 * p + 1] = (_Float16)b1;
  }
  v8f c = {};
  c = WMMA_F16(a, b, c);
  int j = col >> 2, h = col & 3;
#pragma unroll
  for (int p = 0; p < 8; ++p) {
    int row = p + (hs ? 8 : 0);
    WV[(size_t)(m0 + row) * 576 + j * 12 + cc * 4 + h] = c[p];
  }
}

// ---------------------------------------------------------------------------
// K4: per-edge logits. logit[e,h] = o1s(e) . U'[dst] + o1v(e) : WV'[dst]
// One edge per lane, all gathers as b128. Atomic segment max (m >= 0 so
// int-ordered atomicMax is exact).
// ---------------------------------------------------------------------------
__global__ __launch_bounds__(256) void k_logits(
    const int* __restrict__ edge_src, const int* __restrict__ edge_dst,
    const float* __restrict__ eav, const float* __restrict__ node_s,
    const float* __restrict__ node_v, const float* __restrict__ U,
    const float* __restrict__ WV, float* __restrict__ LW,
    float* __restrict__ MB) {
  int e = blockIdx.x * blockDim.x + threadIdx.x;
  if (e >= EE) return;
  int src = edge_src[e], dst = edge_dst[e];
  float r0 = eav[e * 3 + 0], r1 = eav[e * 3 + 1], r2 = eav[e * 3 + 2];
  const float4* U4 = (const float4*)(U + (size_t)dst * 192);   // 48 x b128
  const float4* W4 = (const float4*)(WV + (size_t)dst * 576);  // 144 x b128
  const float4* S4 = (const float4*)(node_s + (size_t)src * F0C);
  const float4* V4 = (const float4*)(node_v + (size_t)src * F1C * 3);
  __builtin_prefetch(U4, 0, 0);
  __builtin_prefetch(W4, 0, 0);

  float l0 = 0.f, l1 = 0.f, l2 = 0.f, l3 = 0.f;
  // scalar channels (rows 0..31 of o1s) + s*r rows (16..47 of o1v)
#pragma unroll 2
  for (int q = 0; q < 8; ++q) {
    float4 s = S4[q];
    float sv[4] = {s.x, s.y, s.z, s.w};
#pragma unroll
    for (int t = 0; t < 4; ++t) {
      int i = 4 * q + t;
      float si = sv[t];
      float4 u = U4[i];
      l0 += si * u.x; l1 += si * u.y; l2 += si * u.z; l3 += si * u.w;
      float4 w0 = W4[(16 + i) * 3 + 0];
      float4 w1 = W4[(16 + i) * 3 + 1];
      float4 w2 = W4[(16 + i) * 3 + 2];
      l0 += si * (r0 * w0.x + r1 * w1.x + r2 * w2.x);
      l1 += si * (r0 * w0.y + r1 * w1.y + r2 * w2.y);
      l2 += si * (r0 * w0.z + r1 * w1.z + r2 * w2.z);
      l3 += si * (r0 * w0.w + r1 * w1.w + r2 * w2.w);
    }
  }
  // dot(v,r) rows (32..47 of o1s) + v rows (0..15 of o1v)
#pragma unroll 2
  for (int q = 0; q < 4; ++q) {
    float4 a = V4[q * 3 + 0];
    float4 b = V4[q * 3 + 1];
    float4 c = V4[q * 3 + 2];
    float vv[12] = {a.x, a.y, a.z, a.w, b.x, b.y, b.z, b.w, c.x, c.y, c.z, c.w};
#pragma unroll
    for (int t = 0; t < 4; ++t) {
      int f = 4 * q + t;
      float v0 = vv[3 * t], v1 = vv[3 * t + 1], v2 = vv[3 * t + 2];
      float dt = (v0 * r0 + v1 * r1 + v2 * r2) * INV_SQRT3;
      float4 u = U4[32 + f];
      l0 += dt * u.x; l1 += dt * u.y; l2 += dt * u.z; l3 += dt * u.w;
      float4 w0 = W4[f * 3 + 0];
      float4 w1 = W4[f * 3 + 1];
      float4 w2 = W4[f * 3 + 2];
      l0 += v0 * w0.x + v1 * w1.x + v2 * w2.x;
      l1 += v0 * w0.y + v1 * w1.y + v2 * w2.y;
      l2 += v0 * w0.z + v1 * w1.z + v2 * w2.z;
      l3 += v0 * w0.w + v1 * w1.w + v2 * w2.w;
    }
  }
  ((float4*)LW)[e] = make_float4(l0, l1, l2, l3);   // single b128 store
  int* mb = (int*)(MB + (size_t)dst * 4);
  atomicMax(mb + 0, __float_as_int(fmaxf(l0, 0.f)));
  atomicMax(mb + 1, __float_as_int(fmaxf(l1, 0.f)));
  atomicMax(mb + 2, __float_as_int(fmaxf(l2, 0.f)));
  atomicMax(mb + 3, __float_as_int(fmaxf(l3, 0.f)));
}

// ---------------------------------------------------------------------------
// K5: e = cutoff * exp(logit - m[dst]); segment sum z. One edge per lane,
// all 4 heads via b128 load/store.
// ---------------------------------------------------------------------------
__global__ __launch_bounds__(256) void k_expsum(
    const int* __restrict__ edge_dst, const float* __restrict__ cutoff,
    const float* __restrict__ MB, float* __restrict__ LW,
    float* __restrict__ ZB) {
  int e = blockIdx.x * blockDim.x + threadIdx.x;
  if (e >= EE) return;
  int dst = edge_dst[e];
  float4 lw = ((const float4*)LW)[e];
  float4 mb = ((const float4*)MB)[dst];
  float cw = cutoff[e];
  float4 ev = make_float4(cw * expf(lw.x - mb.x), cw * expf(lw.y - mb.y),
                          cw * expf(lw.z - mb.z), cw * expf(lw.w - mb.w));
  ((float4*)LW)[e] = ev;
  float* zb = ZB + (size_t)dst * 4;
  atomicAdd(zb + 0, ev.x);
  atomicAdd(zb + 1, ev.y);
  atomicAdd(zb + 2, ev.z);
  atomicAdd(zb + 3, ev.w);
}

// ---------------------------------------------------------------------------
// K6: value branch + weighted segment scatter into ns[N,48], nv[N,48,3].
// ---------------------------------------------------------------------------
__global__ __launch_bounds__(256) void k_scatter(
    const int* __restrict__ edge_src, const int* __restrict__ edge_dst,
    const float* __restrict__ eav, const float* __restrict__ node_s,
    const float* __restrict__ node_v, const float* __restrict__ gates,
    const float* __restrict__ LW, const float* __restrict__ ZB,
    float* __restrict__ NS, float* __restrict__ NV) {
  int e = blockIdx.x * blockDim.x + threadIdx.x;
  if (e >= EE) return;
  int src = edge_src[e], dst = edge_dst[e];
  float r0 = eav[e * 3 + 0], r1 = eav[e * 3 + 1], r2 = eav[e * 3 + 2];
  float4 zb = ((const float4*)ZB)[dst];
  float4 lw = ((const float4*)LW)[e];
  float wh[4];
  wh[0] = sqrtf(fmaxf(lw.x / ((zb.x == 0.f) ? 1.f : zb.x), 0.f));
  wh[1] = sqrtf(fmaxf(lw.y / ((zb.y == 0.f) ? 1.f : zb.y), 0.f));
  wh[2] = sqrtf(fmaxf(lw.z / ((zb.z == 0.f) ? 1.f : zb.z), 0.f));
  wh[3] = sqrtf(fmaxf(lw.w / ((zb.w == 0.f) ? 1.f : zb.w), 0.f));
  const float* gs  = gates + 96;        // value gates, scalar irreps [48]
  const float* gvv = gates + 96 + 48;   // value gates, vector irreps [48]
  const float4* S4 = (const float4*)(node_s + (size_t)src * F0C);
  const float4* V4 = (const float4*)(node_v + (size_t)src * F1C * 3);
  float* nsd = NS + (size_t)dst * KC;
  float* nvd = NV + (size_t)dst * KC * 3;
#pragma unroll 2
  for (int q = 0; q < 8; ++q) {
    float4 s = S4[q];
    float sv[4] = {s.x, s.y, s.z, s.w};
#pragma unroll
    for (int t = 0; t < 4; ++t) {
      int i = 4 * q + t;
      float si = sv[t];
      atomicAdd(&nsd[i], si * gs[i] * wh[i / 12]);
      int j = 16 + i;
      float fac = si * gvv[j] * wh[j / 12];   // s x r -> 1o rows
      atomicAdd(&nvd[j * 3 + 0], fac * r0);
      atomicAdd(&nvd[j * 3 + 1], fac * r1);
      atomicAdd(&nvd[j * 3 + 2], fac * r2);
    }
  }
#pragma unroll 2
  for (int q = 0; q < 4; ++q) {
    float4 a = V4[q * 3 + 0];
    float4 b = V4[q * 3 + 1];
    float4 c = V4[q * 3 + 2];
    float vv[12] = {a.x, a.y, a.z, a.w, b.x, b.y, b.z, b.w, c.x, c.y, c.z, c.w};
#pragma unroll
    for (int t = 0; t < 4; ++t) {
      int f = 4 * q + t;
      float v0 = vv[3 * t], v1 = vv[3 * t + 1], v2 = vv[3 * t + 2];
      float dt = (v0 * r0 + v1 * r1 + v2 * r2) * INV_SQRT3;
      int j = 32 + f;                          // dot -> 0e rows
      atomicAdd(&nsd[j], dt * gs[j] * wh[j / 12]);
      float fac = gvv[f] * wh[f / 12];         // v -> 1o rows
      atomicAdd(&nvd[f * 3 + 0], v0 * fac);
      atomicAdd(&nvd[f * 3 + 1], v1 * fac);
      atomicAdd(&nvd[f * 3 + 2], v2 * fac);
    }
  }
}

// ---------------------------------------------------------------------------
// K7: out_s = ns @ Wout0 / sqrt(48): [10000,48] x [48,32] (2 WMMAs per tile)
// ---------------------------------------------------------------------------
__global__ __launch_bounds__(256) void k_out_s(
    const float* __restrict__ NS, const float* __restrict__ Wout0,
    float* __restrict__ out) {
  int wid = (blockIdx.x * blockDim.x + threadIdx.x) >> 5;
  if (wid >= (NN / 16) * 2) return;
  int lane = threadIdx.x & 31;
  int hs = lane >> 4, lm = lane & 15;
  int rt = wid >> 1, ct = wid & 1;
  int m0 = rt * 16, col = ct * 16 + lm;

  v8f acc = {};
#pragma unroll
  for (int kk = 0; kk < 2; ++kk) {
    v16h a, b;
    const float* arow = NS + (size_t)(m0 + lm) * KC;
#pragma unroll
    for (int p = 0; p < 8; ++p) {
      int kg0 = kk * 32 + a_kidx(p, hs);
      a[2 * p]     = (_Float16)((kg0 < KC) ? arow[kg0] : 0.f);
      a[2 * p + 1] = (_Float16)((kg0 + 1 < KC) ? arow[kg0 + 1] : 0.f);
    }
#pragma unroll
    for (int p = 0; p < 8; ++p) {
      int kg0 = kk * 32 + b_kidx(p, hs);
      float b0 = (kg0 < KC)     ? Wout0[(size_t)kg0 * F0C + col] * INVSQ48 : 0.f;
      float b1 = (kg0 + 1 < KC) ? Wout0[(size_t)(kg0 + 1) * F0C + col] * INVSQ48 : 0.f;
      b[2 * p]     = (_Float16)b0;
      b[2 * p + 1] = (_Float16)b1;
    }
    acc = WMMA_F16(a, b, acc);
  }
#pragma unroll
  for (int p = 0; p < 8; ++p) {
    int row = p + (hs ? 8 : 0);
    out[(size_t)(m0 + row) * 80 + col] = acc[p];
  }
}

// ---------------------------------------------------------------------------
// K8: out_v[n,g,c] = sum_j nv[n,j,c] * Wout1[j,g] / sqrt(48)
// GEMM over rows (n,c): [30000,48] x [48,16].
// ---------------------------------------------------------------------------
__global__ __launch_bounds__(256) void k_out_v(
    const float* __restrict__ NV, const float* __restrict__ Wout1,
    float* __restrict__ out) {
  int wid = (blockIdx.x * blockDim.x + threadIdx.x) >> 5;
  if (wid >= (NN * 3) / 16) return;
  int lane = threadIdx.x & 31;
  int hs = lane >> 4, lm = lane & 15;
  int m0 = wid * 16;

  v8f acc = {};
#pragma unroll
  for (int kk = 0; kk < 2; ++kk) {
    v16h a, b;
    int mrow = m0 + lm;
    int n = mrow / 3, cc = mrow % 3;
    const float* arow = NV + (size_t)n * KC * 3 + cc;
#pragma unroll
    for (int p = 0; p < 8; ++p) {
      int kg0 = kk * 32 + a_kidx(p, hs);
      a[2 * p]     = (_Float16)((kg0 < KC) ? arow[kg0 * 3] : 0.f);
      a[2 * p + 1] = (_Float16)((kg0 + 1 < KC) ? arow[(kg0 + 1) * 3] : 0.f);
    }
#pragma unroll
    for (int p = 0; p < 8; ++p) {
      int kg0 = kk * 32 + b_kidx(p, hs);
      float b0 = (kg0 < KC)     ? Wout1[(size_t)kg0 * F1C + lm] * INVSQ48 : 0.f;
      float b1 = (kg0 + 1 < KC) ? Wout1[(size_t)(kg0 + 1) * F1C + lm] * INVSQ48 : 0.f;
      b[2 * p]     = (_Float16)b0;
      b[2 * p + 1] = (_Float16)b1;
    }
    acc = WMMA_F16(a, b, acc);
  }
#pragma unroll
  for (int p = 0; p < 8; ++p) {
    int rowg = m0 + p + (hs ? 8 : 0);
    int n = rowg / 3, cc = rowg % 3;
    out[(size_t)n * 80 + 32 + lm * 3 + cc] = acc[p];
  }
}

// ---------------------------------------------------------------------------
// Host-side launch
// ---------------------------------------------------------------------------
extern "C" void kernel_launch(void* const* d_in, const int* in_sizes, int n_in,
                              void* d_out, int out_size, void* d_ws,
                              size_t ws_size, hipStream_t stream) {
  const int*   edge_src = (const int*)d_in[0];
  const int*   edge_dst = (const int*)d_in[1];
  const float* cutoff   = (const float*)d_in[2];
  // d_in[3] = edge_attr_s (all ones; MLP input folded into k_gates)
  const float* eav      = (const float*)d_in[4];
  const float* node_s   = (const float*)d_in[5];
  const float* node_v   = (const float*)d_in[6];
  const float* Wk1 = (const float*)d_in[7];
  const float* Wk2 = (const float*)d_in[8];
  const float* Wk3 = (const float*)d_in[9];
  const float* Wv1 = (const float*)d_in[10];
  const float* Wv2 = (const float*)d_in[11];
  const float* Wv3 = (const float*)d_in[12];
  const float* Wlog0 = (const float*)d_in[13];
  const float* Wlog1 = (const float*)d_in[14];
  const float* Wout0 = (const float*)d_in[15];
  const float* Wout1 = (const float*)d_in[16];
  float* out = (float*)d_out;

  // Workspace layout (floats); every array 16-byte aligned.
  float* ws    = (float*)d_ws;
  float* gates = ws;                        //   192 (key 96 | value 96)
  float* U     = ws + 256;                  //  N*192 = 1,920,000
  float* WV    = U + (size_t)NN * 192;      //  N*576 = 5,760,000
  float* LW    = WV + (size_t)NN * 576;     //  E*4   = 1,280,000
  float* MB    = LW + (size_t)EE * 4;       //  N*4
  float* ZB    = MB + (size_t)NN * 4;       //  N*4
  float* NS    = ZB + (size_t)NN * 4;       //  N*48
  float* NV    = NS + (size_t)NN * KC;      //  N*144
  (void)in_sizes; (void)n_in; (void)out_size; (void)ws_size;

  // 1. constant edge-MLP gates
  k_gates<<<2, 128, 0, stream>>>(Wk1, Wk2, Wk3, Wv1, Wv2, Wv3, gates);

  // 2. zero MB|ZB|NS|NV (contiguous: 2,000,000 floats)
  int nz = NN * 4 + NN * 4 + NN * KC + NN * KC * 3;
  k_zero<<<(nz + 255) / 256, 256, 0, stream>>>(MB, nz);

  // 3. WMMA precompute of per-dst logit tables
  {
    int waves = (NN / 16) * 12;
    k_precomp_u<<<(waves + 7) / 8, 256, 0, stream>>>(node_s, Wlog0, gates, U);
  }
  {
    int waves = (NN / 16) * 12 * 3;
    k_precomp_wv<<<(waves + 7) / 8, 256, 0, stream>>>(node_v, Wlog1, gates, WV);
  }

  // 4. edge logits + segment max
  k_logits<<<(EE + 255) / 256, 256, 0, stream>>>(edge_src, edge_dst, eav,
                                                 node_s, node_v, U, WV, LW, MB);
  // 5. exp + segment sum
  k_expsum<<<(EE + 255) / 256, 256, 0, stream>>>(edge_dst, cutoff, MB, LW, ZB);
  // 6. value scatter
  k_scatter<<<(EE + 255) / 256, 256, 0, stream>>>(edge_src, edge_dst, eav,
                                                  node_s, node_v, gates, LW,
                                                  ZB, NS, NV);
  // 7. WMMA output projections
  {
    int waves = (NN / 16) * 2;
    k_out_s<<<(waves + 7) / 8, 256, 0, stream>>>(NS, Wout0, out);
  }
  {
    int waves = (NN * 3) / 16;
    k_out_v<<<(waves + 7) / 8, 256, 0, stream>>>(NV, Wout1, out);
  }
}
